// GraphAggregationLayer_43267500540215
// MI455X (gfx1250) — matrix operand order repaired and checked by reading, not compiled, not run on hardware
//
#include <hip/hip_runtime.h>
#include <hip/hip_bf16.h>

typedef float  v2f   __attribute__((ext_vector_type(2)));
typedef float  v4f   __attribute__((ext_vector_type(4)));
typedef float  v8f   __attribute__((ext_vector_type(8)));
typedef int    v4i   __attribute__((ext_vector_type(4)));
typedef __bf16 v8bf  __attribute__((ext_vector_type(8)));
typedef __bf16 v16bf __attribute__((ext_vector_type(16)));

#define Bn 8192
#define Dn 256
#define NEG_SLOPE 0.2f
#define LN_EPS 1e-5f

// ---------------------------------------------------------------------------
// Kernel 1: Wh = h @ W^T in f32 WMMA (16x16x4), store transposed bf16 WhT[D][B]
// One wave computes one 16x16 tile. Lane mapping (wave32):
//   M/N = lane&15, kg = lane>>4.
// A (16x4 f32): lane holds A[M][2*kg + {0,1}]      -> contiguous b64 from h
// B (4x16 f32): lane holds B[2*kg + {0,1}][N]=W[N][k] -> contiguous b64 from W
// C (16x16 f32): VGPR v, lane -> row M = v + 8*kg, col N = lane&15
// Store: rows i0+8*kg+0..7 at column d0+N of WhT -> 8 consecutive bf16 = b128.
// ---------------------------------------------------------------------------
__global__ void __launch_bounds__(256) k_wh(const float* __restrict__ h,
                                            const float* __restrict__ W,
                                            __bf16* __restrict__ WhT) {
  const int wave = (blockIdx.x * blockDim.x + threadIdx.x) >> 5;  // 0..8191
  const int lane = threadIdx.x & 31;
  const int ib = wave >> 4;   // 0..511  (i block)
  const int db = wave & 15;   // 0..15   (d block)
  const int i0 = ib * 16, d0 = db * 16;
  const int M  = lane & 15;   // also N
  const int kg = lane >> 4;

  const float* ha = h + (size_t)(i0 + M) * Dn + 2 * kg;
  const float* wb = W + (size_t)(d0 + M) * Dn + 2 * kg;  // B[k][N] = W[N][k]

  v8f c = {};
#pragma unroll 4
  for (int k0 = 0; k0 < Dn; k0 += 4) {
    v2f a = *(const v2f*)(ha + k0);
    v2f b = *(const v2f*)(wb + k0);
    c = __builtin_amdgcn_wmma_f32_16x16x4_f32(false, a, false, b,
                                              (short)0, c, false, false);
  }

  v8bf o;
#pragma unroll
  for (int v = 0; v < 8; ++v) o[v] = (__bf16)c[v];
  // WhT[(d0+N)*Bn + i0 + 8*kg .. +7]  (16-byte aligned)
  *(v8bf*)(WhT + (size_t)(d0 + M) * Bn + i0 + 8 * kg) = o;
}

// ---------------------------------------------------------------------------
// Kernel 2: si = Wh@a1, sj = Wh@a2, reading WhT columns (coalesced over i).
// ---------------------------------------------------------------------------
__global__ void __launch_bounds__(256) k_sisj(const __bf16* __restrict__ WhT,
                                              const float* __restrict__ a1,
                                              const float* __restrict__ a2,
                                              float* __restrict__ si,
                                              float* __restrict__ sj) {
  const int i = blockIdx.x * blockDim.x + threadIdx.x;  // 0..8191
  float s1 = 0.f, s2 = 0.f;
#pragma unroll 4
  for (int d = 0; d < Dn; ++d) {
    float w = (float)WhT[(size_t)d * Bn + i];
    s1 = fmaf(w, a1[d], s1);
    s2 = fmaf(w, a2[d], s2);
  }
  si[i] = s1;
  sj[i] = s2;
}

// ---------------------------------------------------------------------------
// Kernel 3: fused masked-softmax attention + aggregation via bf16 WMMA.
// One wave owns a 16-row block (ib) and a 128-wide d half (dh).
// Single pass (no max subtraction; e ~ N(0,1)):
//   p_ij = adj ? exp(lrelu(si+sj)) : 0 ; Z_i = sum_j p ; acc = P @ Wh
// A-matrix 16x32 bf16 register layout (lane L: M=L&15, kg=L>>4):
//   elems 0..15 -> K = kg*16 + 0..15
// B-matrix 32x16 bf16: lane holds col N=L&15, K = kg*16 + 0..15 -> 32B from WhT.
// ---------------------------------------------------------------------------
__global__ void __launch_bounds__(256, 1) k_attn(const __bf16* __restrict__ WhT,
                                                 const int*   __restrict__ adj,
                                                 const float* __restrict__ si,
                                                 const float* __restrict__ sj,
                                                 const float* __restrict__ h,
                                                 float* __restrict__ out) {
  const int g    = (blockIdx.x * blockDim.x + threadIdx.x) >> 5;  // 0..1023
  const int lane = threadIdx.x & 31;
  const int ib = g >> 1;            // 0..511
  const int dh = g & 1;             // 0..1  (d half; pair shares adj rows -> L2)
  const int i0 = ib * 16, dbase = dh * 128;
  const int M  = lane & 15;
  const int kg = lane >> 4;
  const int ra = kg * 16;           // K run offset within a 32-j block

  const float si_m   = si[i0 + M];
  const int* adjrow  = adj + (size_t)(i0 + M) * Bn;

  const __bf16* bptr[8];
#pragma unroll
  for (int dc = 0; dc < 8; ++dc)
    bptr[dc] = WhT + (size_t)(dbase + dc * 16 + M) * Bn + kg * 16;

  v8f acc[8];
#pragma unroll
  for (int dc = 0; dc < 8; ++dc) acc[dc] = (v8f){};
  float zacc = 0.f;

  for (int j0 = 0; j0 < Bn; j0 += 32) {
    // stage the 16 (sj, adj) pairs for this lane's K slots
    float sv[16];
    int   mv[16];
    *(v4f*)(sv + 0)  = *(const v4f*)(sj + j0 + ra);
    *(v4f*)(sv + 4)  = *(const v4f*)(sj + j0 + ra + 4);
    *(v4f*)(sv + 8)  = *(const v4f*)(sj + j0 + ra + 8);
    *(v4f*)(sv + 12) = *(const v4f*)(sj + j0 + ra + 12);
    *(v4i*)(mv + 0)  = *(const v4i*)(adjrow + j0 + ra);
    *(v4i*)(mv + 4)  = *(const v4i*)(adjrow + j0 + ra + 4);
    *(v4i*)(mv + 8)  = *(const v4i*)(adjrow + j0 + ra + 8);
    *(v4i*)(mv + 12) = *(const v4i*)(adjrow + j0 + ra + 12);
    if (j0 + 32 < Bn)
      __builtin_prefetch(adjrow + j0 + 32 + ra, 0, 1);  // global_prefetch_b8

    v16bf A;
#pragma unroll
    for (int t = 0; t < 16; ++t) {
      float e = si_m + sv[t];
      e = (e >= 0.f) ? e : NEG_SLOPE * e;                // LeakyReLU(0.2)
      float p = (mv[t] != 0) ? __expf(e) : 0.f;          // mask == exp(-1e9)
      zacc += p;
      A[t] = (__bf16)p;
    }

#pragma unroll
    for (int dc = 0; dc < 8; ++dc) {
      v16bf Bv = *(const v16bf*)(bptr[dc] + j0);
      acc[dc] = __builtin_amdgcn_wmma_f32_16x16x32_bf16(false, A, false, Bv,
                                                        (short)0, acc[dc],
                                                        false, false);
    }
  }

  // Z per row: lane L covers half the K's for row M; partner is lane^16.
  float zrow = zacc + __shfl_xor(zacc, 16, 32);  // valid for row (lane&15)

#pragma unroll
  for (int v = 0; v < 8; ++v) {
    const int row = i0 + v + 8 * kg;                   // C layout: M = v + 8*kg
    float z   = __shfl(zrow, v + 8 * kg, 32);
    float inv = (z > 0.f) ? 1.f / z : 0.f;             // all-masked row -> 0
#pragma unroll
    for (int dc = 0; dc < 8; ++dc) {
      float a = acc[dc][v] * inv;                      // alpha @ Wh
      float e = (a > 0.f) ? a : (__expf(a) - 1.f);     // ELU
      size_t idx = (size_t)row * Dn + dbase + dc * 16 + M;
      out[idx] = h[idx] + e;                           // pre-LN residual
    }
  }
}

// ---------------------------------------------------------------------------
// Kernel 4: in-place LayerNorm over D=256, one wave per row (8 elems/lane).
// ---------------------------------------------------------------------------
__global__ void __launch_bounds__(256) k_ln(float* __restrict__ out,
                                            const float* __restrict__ gamma,
                                            const float* __restrict__ beta) {
  const int row  = (blockIdx.x * blockDim.x + threadIdx.x) >> 5;  // 0..8191
  const int lane = threadIdx.x & 31;
  float* r = out + (size_t)row * Dn;

  float x[8];
  float s = 0.f;
#pragma unroll
  for (int t = 0; t < 8; ++t) { x[t] = r[lane + 32 * t]; s += x[t]; }
#pragma unroll
  for (int o = 16; o >= 1; o >>= 1) s += __shfl_xor(s, o, 32);
  const float mu = s * (1.f / (float)Dn);

  float vs = 0.f;
#pragma unroll
  for (int t = 0; t < 8; ++t) { float d = x[t] - mu; vs += d * d; }
#pragma unroll
  for (int o = 16; o >= 1; o >>= 1) vs += __shfl_xor(vs, o, 32);
  const float rstd = rsqrtf(vs * (1.f / (float)Dn) + LN_EPS);

#pragma unroll
  for (int t = 0; t < 8; ++t) {
    int c = lane + 32 * t;
    r[c] = (x[t] - mu) * rstd * gamma[c] + beta[c];
  }
}

// ---------------------------------------------------------------------------
extern "C" void kernel_launch(void* const* d_in, const int* in_sizes, int n_in,
                              void* d_out, int out_size, void* d_ws, size_t ws_size,
                              hipStream_t stream) {
  const float* h     = (const float*)d_in[0];   // [B, D]
  const int*   adj   = (const int*)  d_in[1];   // [B, B]
  const float* W_w   = (const float*)d_in[2];   // [D, D]
  const float* a1    = (const float*)d_in[3];   // [D]
  const float* a2    = (const float*)d_in[4];   // [D]
  const float* gamma = (const float*)d_in[5];   // [D]
  const float* beta  = (const float*)d_in[6];   // [D]
  float* out = (float*)d_out;                   // [B, D] f32

  // workspace layout: WhT bf16 [D][B] (4 MB), si [B], sj [B]
  char* ws = (char*)d_ws;
  __bf16* WhT = (__bf16*)ws;                          // 256*8192*2 = 4 MiB
  float*  si  = (float*)(ws + (size_t)Dn * Bn * 2);
  float*  sj  = si + Bn;

  // 1) Wh via f32 WMMA, stored transposed bf16: 8192 tile-waves, 8 waves/block
  k_wh<<<(Bn / 16) * (Dn / 16) / 8, 256, 0, stream>>>(h, W_w, WhT);
  // 2) attention logits halves
  k_sisj<<<Bn / 256, 256, 0, stream>>>(WhT, a1, a2, si, sj);
  // 3) fused masked softmax + aggregation (bf16 WMMA) + ELU + residual
  k_attn<<<(Bn / 16) * 2 / 8, 256, 0, stream>>>(WhT, adj, si, sj, h, out);
  // 4) in-place LayerNorm
  k_ln<<<Bn / 8, 256, 0, stream>>>(out, gamma, beta);
}
